// Generator_7524782702957
// MI455X (gfx1250) — matrix operand order repaired and checked by reading, not compiled, not run on hardware
//
#include <hip/hip_runtime.h>
#include <hip/hip_bf16.h>

typedef __attribute__((ext_vector_type(16))) __bf16 v16bf;
typedef __attribute__((ext_vector_type(8)))  __bf16 v8bf;
typedef __attribute__((ext_vector_type(8)))  float  v8f;
typedef __attribute__((ext_vector_type(4)))  int    v4i;

#define HIDDEN 1024
#define SEQLEN 256
#define BATCH  256

#define KC      256                  // K-chunk staged per async stage
#define NCHUNK  (HIDDEN / KC)        // 4
#define RS      (KC + 8)             // LDS row stride in bf16 elems (16B pad)
#define BUFELEM (64 * RS)            // one staging buffer: 64 rows

// ---- CDNA5 async global->LDS copy (ASYNCcnt-tracked) ----------------------
// Builtin signature (from clang diagnostic): (v4i AS1*, v4i AS3*, imm, imm)
#if defined(__has_builtin) && __has_builtin(__builtin_amdgcn_global_load_async_to_lds_b128)
#define ASYNC_COPY16(gsrc, ldst)                                              \
    __builtin_amdgcn_global_load_async_to_lds_b128(                           \
        (__attribute__((address_space(1))) v4i*)(gsrc),                       \
        (__attribute__((address_space(3))) v4i*)(ldst), 0, 0)
#else
#define ASYNC_COPY16(gsrc, ldst)                                              \
    asm volatile("global_load_async_to_lds_b128 %0, %1, off"                  \
                 :: "v"((unsigned)(unsigned long long)(ldst)),                \
                    "v"((const void*)(gsrc)) : "memory")
#endif

#if defined(__has_builtin) && __has_builtin(__builtin_amdgcn_s_wait_asynccnt)
#define WAIT_ASYNC(n) __builtin_amdgcn_s_wait_asynccnt(n)
#else
#define WAIT_ASYNC(n) asm volatile("s_wait_asynccnt %0" :: "i"(n) : "memory")
#endif

// ---------------------------------------------------------------------------
// Init: convert W_hh to bf16 (L2-resident), zero h0/c, pre-fill out with b_out.
// ---------------------------------------------------------------------------
__global__ __launch_bounds__(256) void lstm_init_kernel(
    const float* __restrict__ Whh,
    const float* __restrict__ b_out,
    __bf16* __restrict__ Whh_bf,
    __bf16* __restrict__ h0,
    float*  __restrict__ c,
    float*  __restrict__ out)
{
    const int W  = 4 * HIDDEN * HIDDEN;
    const int HC = BATCH * HIDDEN;
    const int OT = BATCH * SEQLEN;
    const int stride = gridDim.x * blockDim.x;
    const float bo = b_out[0];

    for (int i = blockIdx.x * blockDim.x + threadIdx.x; i < W; i += stride)
        Whh_bf[i] = (__bf16)Whh[i];
    for (int i = blockIdx.x * blockDim.x + threadIdx.x; i < HC; i += stride) {
        h0[i] = (__bf16)0.0f;
        c[i]  = 0.0f;
    }
    for (int i = blockIdx.x * blockDim.x + threadIdx.x; i < OT; i += stride)
        out[i] = bo;
}

// ---------------------------------------------------------------------------
// One LSTM timestep. Grid: x = H/16 (64), y = B/128 (2). Block: 256 thr / 8 waves.
// Weight tile (4 gates x 16 rows x H) staged into LDS via async DMA,
// double-buffered in KC-column chunks; B fragments read with ds_load_b128.
// ---------------------------------------------------------------------------
__global__ __launch_bounds__(256) void lstm_step_kernel(
    const __bf16* __restrict__ hprev,   // [B,H] bf16
    __bf16*       __restrict__ hnext,   // [B,H] bf16
    float*        __restrict__ c,       // [B,H] f32
    const __bf16* __restrict__ Whh,     // [4H,H] bf16
    const float*  __restrict__ x,       // [B,T]
    const float*  __restrict__ W_ih,    // [4H]
    const float*  __restrict__ b_ih,    // [4H]
    const float*  __restrict__ b_hh,    // [4H]
    const float*  __restrict__ W_out,   // [H]
    float*        __restrict__ out,     // [B,T]
    int t)
{
    __shared__ __align__(16) __bf16 wlds[2 * BUFELEM];  // 66 KB weight staging
    __shared__ float xs[128];
    __shared__ float wih_s[64];
    __shared__ float bias_s[64];
    __shared__ float wout_s[16];

    const int H     = HIDDEN;
    const int jb    = blockIdx.x;        // 0..63
    const int mb    = blockIdx.y;        // 0..1
    const int tid   = threadIdx.x;
    const int lane  = tid & 31;
    const int wv    = tid >> 5;          // wave 0..7
    const int j0    = jb * 16;
    const int mbase = mb * 128;
    const int m0    = mbase + wv * 16;
    const int ln15  = lane & 15;
    const int hi    = lane >> 4;

    // ---- stage chunk 0 of the weight tile (async DMA, 8 rows per issue) ----
    // Row r (0..63): gate g=r>>4, col j0+(r&15); 512B/row = 32 lanes x 16B.
    {
        const int k0 = 0;
#pragma unroll
        for (int issue = 0; issue < 8; ++issue) {
            const int row = issue * 8 + wv;
            const int g = row >> 4, jr = row & 15;
            const __bf16* gsrc =
                Whh + (size_t)(g * H + j0 + jr) * H + k0 + lane * 8;
            __bf16* ldst = &wlds[row * RS + lane * 8];
            ASYNC_COPY16(gsrc, ldst);
        }
    }

    // small parameter tiles (overlap with DMA)
    if (tid < 128) xs[tid] = x[(mbase + tid) * SEQLEN + t];
    if (tid < 64) {
        int g   = tid >> 4;
        int row = g * H + j0 + (tid & 15);
        wih_s[tid]  = W_ih[row];
        bias_s[tid] = b_ih[row] + b_hh[row];
    }
    if (tid < 16) wout_s[tid] = W_out[j0 + tid];

    // A fragment base (16x32 bf16 per K-step, two b128 per lane from global)
    const __bf16* aptr = hprev + (size_t)(m0 + ln15) * H + (hi ? 8 : 0);

    v8f acc[4] = {};

    for (int ch = 0; ch < NCHUNK; ++ch) {
        const __bf16* wbuf = &wlds[(ch & 1) * BUFELEM];

        // prefetch next chunk into the other buffer
        if (ch + 1 < NCHUNK) {
            __bf16* nbuf = &wlds[((ch + 1) & 1) * BUFELEM];
            const int k0 = (ch + 1) * KC;
#pragma unroll
            for (int issue = 0; issue < 8; ++issue) {
                const int row = issue * 8 + wv;
                const int g = row >> 4, jr = row & 15;
                const __bf16* gsrc =
                    Whh + (size_t)(g * H + j0 + jr) * H + k0 + lane * 8;
                __bf16* ldst = &nbuf[row * RS + lane * 8];
                ASYNC_COPY16(gsrc, ldst);
            }
            WAIT_ASYNC(8);   // current chunk's 8 copies done; next still in flight
        } else {
            WAIT_ASYNC(0);
        }
        __syncthreads();     // LDS visibility across waves

        // ---- compute KC columns: KC/32 K-steps of 4 WMMAs ----
#pragma unroll
        for (int ks = 0; ks < KC; ks += 32) {
            const int kg = ch * KC + ks;
            v16bf a;
            {
                v8bf a0 = *(const v8bf*)(aptr + kg);
                v8bf a1 = *(const v8bf*)(aptr + kg + 16);
#pragma unroll
                for (int e = 0; e < 8; ++e) { a[e] = a0[e]; a[8 + e] = a1[e]; }
            }
#pragma unroll
            for (int g = 0; g < 4; ++g) {
                // B frag: lane<16 -> col N=lane, K ks..ks+7 and ks+16..ks+23; lane>=16 -> K +8
                const __bf16* brow =
                    &wbuf[(g * 16 + ln15) * RS + ks + hi * 8];
                v16bf bm;
                v8bf b0 = *(const v8bf*)(brow);
                v8bf b1 = *(const v8bf*)(brow + 16);
#pragma unroll
                for (int e = 0; e < 8; ++e) { bm[e] = b0[e]; bm[8 + e] = b1[e]; }
                acc[g] = __builtin_amdgcn_wmma_f32_16x16x32_bf16(
                    false, a, false, bm, (short)0, acc[g], false, false);
            }
        }
        __syncthreads();     // all waves done reading wbuf before it is re-staged
    }

    // ---- epilogue: lanes 0-15 -> (M=v, N=lane); lanes 16-31 -> (M=v+8) ----
    const int j    = j0 + ln15;
    const int rofs = hi ? 8 : 0;
#pragma unroll
    for (int v = 0; v < 8; ++v) {
        const int brow = m0 + v + rofs;
        const float xv = xs[wv * 16 + v + rofs];
        float zi = acc[0][v] + xv * wih_s[     ln15] + bias_s[     ln15];
        float zf = acc[1][v] + xv * wih_s[16 + ln15] + bias_s[16 + ln15];
        float zg = acc[2][v] + xv * wih_s[32 + ln15] + bias_s[32 + ln15];
        float zo = acc[3][v] + xv * wih_s[48 + ln15] + bias_s[48 + ln15];
        float ig = 1.0f / (1.0f + __expf(-zi));
        float fg = 1.0f / (1.0f + __expf(-zf));
        float gg = tanhf(zg);
        float og = 1.0f / (1.0f + __expf(-zo));
        const size_t cidx = (size_t)brow * H + j;
        float cn = fg * c[cidx] + ig * gg;
        c[cidx] = cn;
        float hn = og * tanhf(cn);
        hnext[cidx] = (__bf16)hn;
        float r = hn * wout_s[ln15];
        r += __shfl_xor(r, 1, 32);
        r += __shfl_xor(r, 2, 32);
        r += __shfl_xor(r, 4, 32);
        r += __shfl_xor(r, 8, 32);
        if (ln15 == 0) atomicAdd(&out[brow * SEQLEN + t], r);
    }
}

// ---------------------------------------------------------------------------
extern "C" void kernel_launch(void* const* d_in, const int* in_sizes, int n_in,
                              void* d_out, int out_size, void* d_ws, size_t ws_size,
                              hipStream_t stream) {
    const float* x     = (const float*)d_in[0];
    const float* W_ih  = (const float*)d_in[1];
    const float* b_ih  = (const float*)d_in[2];
    const float* W_hh  = (const float*)d_in[3];
    const float* b_hh  = (const float*)d_in[4];
    const float* W_out = (const float*)d_in[5];
    const float* b_out = (const float*)d_in[6];
    float* out = (float*)d_out;

    char* ws = (char*)d_ws;
    __bf16* Whh_bf = (__bf16*)ws;                                    // 8 MB
    __bf16* h0     = (__bf16*)(ws + (size_t)8 * 1024 * 1024);        // 512 KB
    __bf16* h1     = (__bf16*)(ws + (size_t)8 * 1024 * 1024 + 512 * 1024);
    float*  c      = (float*) (ws + (size_t)9 * 1024 * 1024);        // 1 MB

    lstm_init_kernel<<<2048, 256, 0, stream>>>(W_hh, b_out, Whh_bf, h0, c, out);

    for (int t = 0; t < SEQLEN; ++t) {
        const __bf16* hp = (t & 1) ? h1 : h0;
        __bf16*       hn = (t & 1) ? h0 : h1;
        lstm_step_kernel<<<dim3(64, 2), 256, 0, stream>>>(
            hp, hn, c, Whh_bf, x, W_ih, b_ih, b_hh, W_out, out, t);
    }
}
